// iPEPS_81655918232073
// MI455X (gfx1250) — matrix operand order repaired and checked by reading, not compiled, not run on hardware
//
#include <hip/hip_runtime.h>

// ---------------------------------------------------------------------------
// iPEPS CTMRG for MI455X (gfx1250).  All tensor contractions are lowered to a
// bf16-input / f32-accumulate WMMA GEMM (v_wmma_f32_16x16x32_bf16).  LDS tiles
// are stored in WMMA *fragment order* so each lane fetches its whole operand
// fragment with wide ds_load_b128 ops instead of per-element ds_load_u16.
// The truncated SVD is replaced by WMMA-driven subspace iteration+CholeskyQR.
// ---------------------------------------------------------------------------

typedef __attribute__((ext_vector_type(16))) __bf16 v16bf;
typedef __attribute__((ext_vector_type(8)))  float  v8f;

__device__ __forceinline__ unsigned short f2bf(float f) {
  unsigned int x = __float_as_uint(f);
  x += 0x7FFFu + ((x >> 16) & 1u);          // round-to-nearest-even
  return (unsigned short)(x >> 16);
}
__device__ __forceinline__ unsigned int pack2bf(float lo, float hi) {
  return (unsigned int)f2bf(lo) | ((unsigned int)f2bf(hi) << 16);
}

#define BM 128
#define BN 64
#define BK 32

// C[M,N] (+)= opA(A) * opB(B); opA: 0 -> A[m,k], 1 -> A[k,m]
//                              opB: 0 -> B[k,n], 1 -> B[n,k]
// blockIdx.z batches with element strides sA/sB/sC.
// 128 threads = 4 wave32 waves; wave owns a 32x64 output strip (8 WMMA tiles).
__global__ __launch_bounds__(128)
void gemm_bf16_k(const float* __restrict__ Ag, const float* __restrict__ Bg,
                 float* __restrict__ Cg, int M, int N, int K,
                 int lda, int ldb, int ldc, int opA, int opB, int accum,
                 long long sA, long long sB, long long sC)
{
  const float* A = Ag + (size_t)blockIdx.z * sA;
  const float* B = Bg + (size_t)blockIdx.z * sB;
  float*       C = Cg + (size_t)blockIdx.z * sC;

  // fragment-order LDS: [subtile][lane][vgpr-slot] ; lane row = 32 B
  __shared__ __align__(32) unsigned int AsF[8][32][8];   // 8 M-subtiles of 16
  __shared__ __align__(32) unsigned int BsF[4][32][8];   // 4 N-subtiles of 16

  const int tid  = threadIdx.x;
  const int lane = tid & 31;
  const int wave = tid >> 5;                 // 0..3
  const int m0 = blockIdx.y * BM;
  const int n0 = blockIdx.x * BN;

  v8f acc[2][4] = {};                        // 2 M-subtiles x 4 N-subtiles

  for (int k0 = 0; k0 < K; k0 += BK) {
    // ---- stage A tile (128x32) directly in fragment order: 2048 u32 ----
    for (int e = tid; e < 2048; e += 128) {
      int mloc = e >> 4;                     // 0..127
      int kp   = e & 15;                     // K pair index
      int ka   = 2 * kp;                     // even K within tile
      int m = m0 + mloc, k = k0 + ka;
      float v0 = 0.f, v1 = 0.f;
      if (m < M) {
        if (k < K)     v0 = opA ? A[(size_t)k * lda + m]       : A[(size_t)m * lda + k];
        if (k + 1 < K) v1 = opA ? A[(size_t)(k + 1) * lda + m] : A[(size_t)m * lda + k + 1];
      }
      // 16-bit A-matrix 16x32 VGPR layout (ISA 7.12.2)
      int kk   = ka & 15;
      int half = kk >> 3;                    // lane half
      int slot = ((ka >= 16) ? 4 : 0) + ((kk & 7) >> 1);
      AsF[mloc >> 4][half * 16 + (mloc & 15)][slot] = pack2bf(v0, v1);
    }
    // ---- stage B tile (32x64) in fragment order: 1024 u32 ----
    for (int e = tid; e < 1024; e += 128) {
      int kp   = e >> 6;                     // 0..15
      int nloc = e & 63;
      int kb   = 2 * kp;
      int k = k0 + kb, n = n0 + nloc;
      float v0 = 0.f, v1 = 0.f;
      if (n < N) {
        if (k < K)     v0 = opB ? B[(size_t)n * ldb + k]     : B[(size_t)k * ldb + n];
        if (k + 1 < K) v1 = opB ? B[(size_t)n * ldb + k + 1] : B[(size_t)(k + 1) * ldb + n];
      }
      // 16-bit B-matrix 32x16 layout (same K order as A fragment)
      int half = (kb >> 3) & 1;
      int slot = ((kb >> 4) << 2) + ((kb & 7) >> 1);
      BsF[nloc >> 4][half * 16 + (nloc & 15)][slot] = pack2bf(v0, v1);
    }
    // prefetch next K tile (speculative)
    if (k0 + BK < K) {
      const float* pa = opA ? &A[(size_t)(k0 + BK) * lda + m0]
                            : &A[(size_t)m0 * lda + (k0 + BK)];
      const float* pb = opB ? &B[(size_t)n0 * ldb + (k0 + BK)]
                            : &B[(size_t)(k0 + BK) * ldb + n0];
      __builtin_prefetch(pa + tid, 0, 1);
      __builtin_prefetch(pb + tid, 0, 1);
    }
    __syncthreads();

    // fragment loads: one 32B vector load each (2 x ds_load_b128)
    v16bf afv0 = *(const v16bf*)&AsF[wave * 2 + 0][lane][0];
    v16bf afv1 = *(const v16bf*)&AsF[wave * 2 + 1][lane][0];
#pragma unroll
    for (int nt = 0; nt < 4; ++nt) {
      v16bf bfv = *(const v16bf*)&BsF[nt][lane][0];
      acc[0][nt] = __builtin_amdgcn_wmma_f32_16x16x32_bf16(
          false, afv0, false, bfv, (short)0, acc[0][nt], false, false);
      acc[1][nt] = __builtin_amdgcn_wmma_f32_16x16x32_bf16(
          false, afv1, false, bfv, (short)0, acc[1][nt], false, false);
    }
    __syncthreads();
  }

  // C/D layout: lane -> N = lane&15 ; VGPR r -> M = r + 8*(lane>>4)
  const int nloc  = lane & 15;
  const int mhalf = (lane >> 4) * 8;
#pragma unroll
  for (int mi = 0; mi < 2; ++mi) {
#pragma unroll
    for (int nt = 0; nt < 4; ++nt) {
      int n = n0 + nt * 16 + nloc;
#pragma unroll
      for (int r = 0; r < 8; ++r) {
        int m = m0 + (wave * 2 + mi) * 16 + mhalf + r;
        if (m < M && n < N) {
          float v = acc[mi][nt][r];
          if (accum) v += C[(size_t)m * ldc + n];
          C[(size_t)m * ldc + n] = v;
        }
      }
    }
  }
}

// ------------------------------ utility kernels ----------------------------

__global__ void zero_k(float* p, long long n) {
  for (long long i = blockIdx.x * 256LL + threadIdx.x; i < n;
       i += (long long)gridDim.x * 256LL) p[i] = 0.f;
}
__global__ void copy_k(float* d, const float* s, long long n) {
  for (long long i = blockIdx.x * 256LL + threadIdx.x; i < n;
       i += (long long)gridDim.x * 256LL) d[i] = s[i];
}
__global__ void sumsq_k(const float* x, long long n, float* acc) {
  __shared__ float sh[256];
  float s = 0.f;
  for (long long i = blockIdx.x * 256LL + threadIdx.x; i < n;
       i += (long long)gridDim.x * 256LL) { float v = x[i]; s += v * v; }
  sh[threadIdx.x] = s; __syncthreads();
  for (int o = 128; o > 0; o >>= 1) {
    if (threadIdx.x < o) sh[threadIdx.x] += sh[threadIdx.x + o];
    __syncthreads();
  }
  if (threadIdx.x == 0) atomicAdd(acc, sh[0]);
}
__global__ void scale_rnorm_k(float* x, long long n, const float* acc) {
  float r = rsqrtf(fmaxf(*acc, 1e-30f));
  for (long long i = blockIdx.x * 256LL + threadIdx.x; i < n;
       i += (long long)gridDim.x * 256LL) x[i] *= r;
}

// generic 4D permute: out = in.transpose(p)   (numpy semantics)
__global__ void perm4_k(const float* __restrict__ in, float* __restrict__ out,
                        int s0, int s1, int s2, int s3,
                        int p0, int p1, int p2, int p3) {
  int sd[4] = {s0, s1, s2, s3};
  long long st[4];
  st[3] = 1; st[2] = s3; st[1] = (long long)s2 * s3; st[0] = (long long)s1 * s2 * s3;
  int pp[4] = {p0, p1, p2, p3};
  int e1 = sd[p1], e2 = sd[p2], e3 = sd[p3];
  long long total = (long long)s0 * s1 * s2 * s3;
  for (long long i = blockIdx.x * 256LL + threadIdx.x; i < total;
       i += (long long)gridDim.x * 256LL) {
    long long t = i;
    int o3 = (int)(t % e3); t /= e3;
    int o2 = (int)(t % e2); t /= e2;
    int o1 = (int)(t % e1); t /= e1;
    int o0 = (int)t;
    out[i] = in[o0 * st[pp[0]] + o1 * st[pp[1]] + o2 * st[pp[2]] + o3 * st[pp[3]]];
  }
}

// out = 0.5*(in + in.transpose(p)) for rank-5, all dims = 4
__global__ void sym5_k(const float* in, float* out,
                       int p0, int p1, int p2, int p3, int p4) {
  int i = blockIdx.x * 256 + threadIdx.x;
  if (i >= 1024) return;
  int o[5]; int t = i;
  for (int k = 4; k >= 0; --k) { o[k] = t & 3; t >>= 2; }
  int pp[5] = {p0, p1, p2, p3, p4};
  const int st[5] = {256, 64, 16, 4, 1};
  int off = 0;
  for (int k = 0; k < 5; ++k) off += o[k] * st[pp[k]];
  out[i] = 0.5f * (in[i] + in[off]);
}

// T8[i,j] = sum_m Am[m,i]*Am[m,j], Am = Asymm.reshape(4,256)
__global__ void tmat_k(const float* As, float* T8) {
  int i = blockIdx.x * 256 + threadIdx.x;
  if (i >= 65536) return;
  int r = i >> 8, c = i & 255;
  float s = 0.f;
  for (int m = 0; m < 4; ++m) s += As[m * 256 + r] * As[m * 256 + c];
  T8[i] = s;
}

// T = T8.transpose(0,4,1,5,2,6,3,7), all 8 dims = 4
__global__ void t8perm_k(const float* in, float* out) {
  int i = blockIdx.x * 256 + threadIdx.x;
  if (i >= 65536) return;
  const int p[8] = {0, 4, 1, 5, 2, 6, 3, 7};
  int o[8]; int t = i;
  for (int k = 7; k >= 0; --k) { o[k] = t & 3; t >>= 2; }
  int off = 0;
  for (int k = 0; k < 8; ++k) off += o[k] << (2 * (7 - p[k]));
  out[i] = in[off];
}

__global__ void cinit_k(const float* T, float* C) {
  int i = blockIdx.x * 256 + threadIdx.x;
  if (i >= 256) return;
  float s = 0.f;
  for (int st = 0; st < 256; ++st) s += T[st * 256 + i];
  C[i] = s;
}
__global__ void einit_k(const float* T, float* E) {
  int i = blockIdx.x * 256 + threadIdx.x;
  if (i >= 4096) return;
  int a = i >> 8, b = (i >> 4) & 15, c = i & 15;
  float s = 0.f;
  for (int t = 0; t < 16; ++t) s += T[((a * 16 + t) * 16 + c) * 16 + b];
  E[i] = s;
}

__global__ void symmat_k(const float* in, float* out, int n, float alpha) {
  long long total = (long long)n * n;
  for (long long i = blockIdx.x * 256LL + threadIdx.x; i < total;
       i += (long long)gridDim.x * 256LL) {
    int r = (int)(i / n), c = (int)(i % n);
    out[i] = alpha * (in[i] + in[(long long)c * n + r]);
  }
}
__global__ void esym_k(const float* in, float* out, int Dn) {
  long long total = (long long)Dn * 16 * Dn;
  for (long long i = blockIdx.x * 256LL + threadIdx.x; i < total;
       i += (long long)gridDim.x * 256LL) {
    int nn = (int)(i / (16LL * Dn));
    int v  = (int)((i / Dn) % 16);
    int m  = (int)(i % Dn);
    out[i] = 0.5f * (in[i] + in[((long long)m * 16 + v) * Dn + nn]);
  }
}

__device__ __forceinline__ unsigned wang(unsigned s) {
  s = (s ^ 61u) ^ (s >> 16); s *= 9u; s ^= s >> 4;
  s *= 0x27d4eb2du; s ^= s >> 15; return s;
}
__global__ void qinit_k(float* Q, long long n) {
  for (long long i = blockIdx.x * 256LL + threadIdx.x; i < n;
       i += (long long)gridDim.x * 256LL)
    Q[i] = (float)wang((unsigned)i) * (1.0f / 4294967296.0f) - 0.5f;
}

// in-place lower Cholesky of symmetric G (n<=256), one workgroup
__global__ void chol_k(float* G, int n) {
  for (int k = 0; k < n; ++k) {
    if (threadIdx.x == 0) G[k * n + k] = sqrtf(fmaxf(G[k * n + k], 1e-20f));
    __syncthreads();
    float dk = G[k * n + k];
    for (int i = k + 1 + threadIdx.x; i < n; i += blockDim.x) G[i * n + k] /= dk;
    __syncthreads();
    int rem = n - k - 1;
    for (int idx = threadIdx.x; idx < rem * rem; idx += blockDim.x) {
      int i = k + 1 + idx / rem;
      int j = k + 1 + idx % rem;
      if (j <= i) G[i * n + j] -= G[i * n + k] * G[j * n + k];
    }
    __syncthreads();
  }
}
// W = L^-1 (lower triangular), one workgroup, one column per thread
__global__ void trinv_k(const float* L, float* W, int n) {
  int j = threadIdx.x;
  if (j >= n) return;
  for (int i = 0; i < n; ++i) W[i * n + j] = 0.f;
  W[j * n + j] = 1.0f / L[j * n + j];
  for (int i = j + 1; i < n; ++i) {
    float s = 0.f;
    for (int k = j; k < i; ++k) s += L[i * n + k] * W[k * n + j];
    W[i * n + j] = -s / L[i * n + i];
  }
}

// Td[(e,a),(f,b),(g,c),(h,d),m,n] = As[m,e,f,g,h]*As[n,a,b,c,d]   (1M elems)
__global__ void td_k(const float* As, float* Td) {
  for (long long ii = blockIdx.x * 256LL + threadIdx.x; ii < 1048576LL;
       ii += (long long)gridDim.x * 256LL) {
    int i = (int)ii;
    int e = (i >> 18) & 3, a = (i >> 16) & 3, f = (i >> 14) & 3, b = (i >> 12) & 3;
    int g = (i >> 10) & 3, c = (i >> 8) & 3, h = (i >> 6) & 3, d = (i >> 4) & 3;
    int m = (i >> 2) & 3,  n = i & 3;
    Td[i] = As[(((m * 4 + e) * 4 + f) * 4 + g) * 4 + h] *
            As[(((n * 4 + a) * 4 + b) * 4 + c) * 4 + d];
  }
}
// Tdp[x2][(s,t)][(x3,m,n)] = Td[t,s,x2,x3,m,n]
__global__ void tdp_k(const float* Td, float* Tdp) {
  for (long long ii = blockIdx.x * 256LL + threadIdx.x; ii < 1048576LL;
       ii += (long long)gridDim.x * 256LL) {
    int i = (int)ii;
    int x2 = i >> 16, s = (i >> 12) & 15, t = (i >> 8) & 15;
    int x3 = (i >> 4) & 15, m = (i >> 2) & 3, n = i & 3;
    Tdp[i] = Td[((((t * 16 + s) * 16 + x2) * 16 + x3) * 16) + m * 4 + n];
  }
}

// Rho4[m,n,mp,np] = sum_{c,x3,c'} EL3[c,(x3,m,n),c'] * EL3[c,(x3,mp,np),c']
__global__ void rho4_k(const float* EL3, float* R4) {
  __shared__ float sh[256];
  int o = blockIdx.x;                        // 256 outputs
  int np = o & 3, mp = (o >> 2) & 3, n = (o >> 4) & 3, m = (o >> 6) & 3;
  float s = 0.f;
  for (int idx = threadIdx.x; idx < 1048576; idx += 256) {
    int c  = idx >> 12;
    int r  = idx & 4095;
    int x3 = r >> 8;
    int c2 = r & 255;
    float a = EL3[((long long)c * 256 + x3 * 16 + m * 4 + n) * 256 + c2];
    float b = EL3[((long long)c * 256 + x3 * 16 + mp * 4 + np) * 256 + c2];
    s += a * b;
  }
  sh[threadIdx.x] = s; __syncthreads();
  for (int w = 128; w > 0; w >>= 1) {
    if (threadIdx.x < w) sh[threadIdx.x] += sh[threadIdx.x + w];
    __syncthreads();
  }
  if (threadIdx.x == 0) R4[o] = sh[0];
}

// loss = tr(Rs@H)/tr(Rs), Rs = 0.5*(Rho5+Rho5^T), Rho5[(m,mp),(n,np)]=R4[m,n,mp,np]
__global__ void loss_k(const float* R4, const float* H, float* out) {
  __shared__ float ra[256], rb[256];
  int i = threadIdx.x >> 4, j = threadIdx.x & 15;
  int m = i >> 2, mp = i & 3, n = j >> 2, np = j & 3;
  float a = R4[((m * 4 + n) * 4 + mp) * 4 + np];
  int m2 = j >> 2, mp2 = j & 3, n2 = i >> 2, np2 = i & 3;
  float b = R4[((m2 * 4 + n2) * 4 + mp2) * 4 + np2];
  float rs = 0.5f * (a + b);
  ra[threadIdx.x] = (i == j) ? rs : 0.f;
  rb[threadIdx.x] = rs * H[j * 16 + i];
  __syncthreads();
  for (int w = 128; w > 0; w >>= 1) {
    if (threadIdx.x < w) {
      ra[threadIdx.x] += ra[threadIdx.x + w];
      rb[threadIdx.x] += rb[threadIdx.x + w];
    }
    __syncthreads();
  }
  if (threadIdx.x == 0) out[0] = rb[0] / ra[0];
}

// ------------------------------ host orchestration -------------------------

static inline int nblk(long long n) {
  long long b = (n + 255) / 256;
  if (b > 32768) b = 32768;
  if (b < 1) b = 1;
  return (int)b;
}
static inline void gemm(hipStream_t s, const float* A, const float* B, float* C,
                        int M, int N, int K, int lda, int ldb, int ldc,
                        int opA, int opB, int accum, int batch = 1,
                        long long sA = 0, long long sB = 0, long long sC = 0) {
  dim3 g((N + BN - 1) / BN, (M + BM - 1) / BM, batch);
  gemm_bf16_k<<<g, 128, 0, s>>>(A, B, C, M, N, K, lda, ldb, ldc,
                                opA, opB, accum, sA, sB, sC);
}
static inline void normalize_buf(hipStream_t s, float* x, long long n, float* acc) {
  zero_k<<<1, 32, 0, s>>>(acc, 1);
  sumsq_k<<<nblk(n), 256, 0, s>>>(x, n, acc);
  scale_rnorm_k<<<nblk(n), 256, 0, s>>>(x, n, acc);
}
static inline void perm4(hipStream_t s, const float* in, float* out,
                         int s0, int s1, int s2, int s3,
                         int p0, int p1, int p2, int p3) {
  long long total = (long long)s0 * s1 * s2 * s3;
  perm4_k<<<nblk(total), 256, 0, s>>>(in, out, s0, s1, s2, s3, p0, p1, p2, p3);
}

// workspace layout (float offsets)
enum : size_t {
  O_T    = 0,
  O_TSW  = O_T + 65536,
  O_C    = O_TSW + 65536,
  O_CN   = O_C + 65536,
  O_G    = O_CN + 65536,
  O_W    = O_G + 65536,
  O_AS   = O_W + 65536,
  O_ATMP = O_AS + 1024,
  O_ACC  = O_ATMP + 1024,
  O_E    = O_ACC + 16,
  O_EN   = O_E + 1048576,
  O_Q    = O_EN + 1048576,
  O_Z    = O_Q + 1048576,
  O_RP   = O_Z + 1048576,
  O_R1   = O_RP + 1048576,
  O_B1   = O_R1 + 1048576,
  O_B2   = O_B1 + 16777216,
  O_B3   = O_B2 + 16777216,
  O_B4   = O_B3 + 16777216,
  O_B5   = O_B4 + 16777216
};

extern "C" void kernel_launch(void* const* d_in, const int* in_sizes, int n_in,
                              void* d_out, int out_size, void* d_ws, size_t ws_size,
                              hipStream_t stream) {
  (void)in_sizes; (void)n_in; (void)out_size; (void)ws_size;
  const float* A_in = (const float*)d_in[0];   // (4,4,4,4,4) f32
  const float* H_in = (const float*)d_in[1];   // (16,16) f32
  float* out = (float*)d_out;
  float* ws  = (float*)d_ws;
  hipStream_t s = stream;

  float* T_   = ws + O_T;   float* TSW_ = ws + O_TSW;
  float* C_   = ws + O_C;   float* CN_  = ws + O_CN;
  float* G_   = ws + O_G;   float* W_   = ws + O_W;
  float* AS_  = ws + O_AS;  float* AT_  = ws + O_ATMP;
  float* ACC_ = ws + O_ACC;
  float* E_   = ws + O_E;   float* EN_  = ws + O_EN;
  float* Q_   = ws + O_Q;   float* Z_   = ws + O_Z;
  float* RP_  = ws + O_RP;  float* R1_  = ws + O_R1;
  float* B1_  = ws + O_B1;  float* B2_  = ws + O_B2;
  float* B3_  = ws + O_B3;  float* B4_  = ws + O_B4;
  float* B5_  = ws + O_B5;

  // --- symmetrize A (4 sequential permutation averages, then normalize) ---
  copy_k<<<4, 256, 0, s>>>(AS_, A_in, 1024);
  sym5_k<<<4, 256, 0, s>>>(AS_, AT_, 0, 4, 3, 2, 1);
  sym5_k<<<4, 256, 0, s>>>(AT_, AS_, 0, 3, 2, 1, 4);
  sym5_k<<<4, 256, 0, s>>>(AS_, AT_, 0, 2, 1, 4, 3);
  sym5_k<<<4, 256, 0, s>>>(AT_, AS_, 0, 1, 4, 3, 2);
  normalize_buf(s, AS_, 1024, ACC_);

  // --- build double-layer T (16,16,16,16), Tsw[s,t,u,v] = T[t,s,u,v] ---
  tmat_k<<<256, 256, 0, s>>>(AS_, CN_);          // T8 scratch in CN_
  t8perm_k<<<256, 256, 0, s>>>(CN_, T_);
  normalize_buf(s, T_, 65536, ACC_);
  perm4(s, T_, TSW_, 16, 16, 16, 16, 1, 0, 2, 3);

  // --- init C, E ---
  cinit_k<<<1, 256, 0, s>>>(T_, C_);
  einit_k<<<16, 256, 0, s>>>(T_, E_);

  const int DT = 16;
  int De = 16;

  // ----------------------------- CTM iterations ----------------------------
  for (int it = 0; it < 6; ++it) {
    const int M4 = De * DT;
    const int Dn = (M4 < 256) ? M4 : 256;

    // Rho1[a,(t,c)] = C @ E
    gemm(s, C_, E_, R1_, De, DT * De, De, De, DT * De, DT * De, 0, 0, 0);
    // Rho2[(t,c),(s,e)] = Rho1^T @ E
    gemm(s, R1_, E_, B1_, DT * De, DT * De, De, DT * De, DT * De, DT * De, 1, 0, 0);
    // Rho2p[(c,e),(t,s)]
    perm4(s, B1_, B2_, DT, De, DT, De, 1, 3, 0, 2);
    // Rho3[(c,e),(u,v)] = Rho2p @ T
    gemm(s, B2_, T_, B3_, De * De, 256, 256, 256, 256, 256, 0, 0, 0);
    // Rho[(c,v),(e,u)]
    perm4(s, B3_, B4_, De, De, DT, DT, 0, 3, 1, 2);
    // RhoS = Rho + Rho^T, normalized
    symmat_k<<<nblk((long long)M4 * M4), 256, 0, s>>>(B4_, B5_, M4, 1.0f);
    normalize_buf(s, B5_, (long long)M4 * M4, ACC_);

    // --- top-256 invariant subspace: block power iteration + CholeskyQR ---
    qinit_k<<<nblk((long long)M4 * Dn), 256, 0, s>>>(Q_, (long long)M4 * Dn);
    for (int pi = 0; pi < 9; ++pi) {
      gemm(s, B5_, Q_, Z_, M4, Dn, M4, M4, Dn, Dn, 0, 0, 0);      // Z = RhoS*Q
      gemm(s, Z_, Z_, G_, Dn, Dn, M4, Dn, Dn, Dn, 1, 0, 0);       // G = Z^T Z
      chol_k<<<1, 256, 0, s>>>(G_, Dn);                           // G -> L
      trinv_k<<<1, 256, 0, s>>>(G_, W_, Dn);                      // W = L^-1
      gemm(s, Z_, W_, Q_, M4, Dn, Dn, Dn, Dn, Dn, 0, 1, 0);       // Q = Z W^T
    }

    // C' = P^T RhoS P (P = Q), then symmetrize + normalize
    gemm(s, B5_, Q_, RP_, M4, Dn, M4, M4, Dn, Dn, 0, 0, 0);
    gemm(s, Q_, RP_, CN_, Dn, Dn, M4, Dn, Dn, Dn, 1, 0, 0);
    symmat_k<<<nblk((long long)Dn * Dn), 256, 0, s>>>(CN_, C_, Dn, 0.5f);
    normalize_buf(s, C_, (long long)Dn * Dn, ACC_);

    // E update: X[(s,e),(t,n)] = E^T(a,(s,e)) @ Pt(a,(t,n))
    gemm(s, E_, Q_, B1_, DT * De, DT * Dn, De, DT * De, DT * Dn, DT * Dn, 1, 0, 0);
    perm4(s, B1_, B2_, DT, De, DT, Dn, 1, 3, 0, 2);   // Xp[(e,n),(s,t)]
    gemm(s, B2_, TSW_, B3_, De * Dn, 256, 256, 256, 256, 256, 0, 0, 0); // Y
    perm4(s, B3_, B4_, De, Dn, DT, DT, 1, 3, 0, 2);   // Yp[(n,v),(e,u)]
    gemm(s, B4_, Q_, EN_, Dn * DT, Dn, De * DT, De * DT, Dn, Dn, 0, 0, 0);
    esym_k<<<nblk((long long)Dn * DT * Dn), 256, 0, s>>>(EN_, E_, Dn);
    normalize_buf(s, E_, (long long)Dn * DT * Dn, ACC_);

    De = Dn;
  }

  // ------------------------------ final loss -------------------------------
  // CE[a,(t,c)] = C @ E
  gemm(s, C_, E_, RP_, 256, 4096, 256, 256, 4096, 4096, 0, 0, 0);
  // EL1[(a,s),(t,c)] = E @ CE
  gemm(s, E_, RP_, B1_, 4096, 4096, 256, 256, 4096, 4096, 0, 0, 0);
  // EL1p[(a,c),(s,t)]
  perm4(s, B1_, B2_, 256, 16, 16, 256, 0, 3, 1, 2);
  // Td / Tdp (reuse Q_, Z_ which are free now)
  td_k<<<4096, 256, 0, s>>>(AS_, Q_);
  tdp_k<<<4096, 256, 0, s>>>(Q_, Z_);
  // EL3[c,(x3,m,n),c'] accumulated over x2 without the 1GB EL2 intermediate
  zero_k<<<nblk(16777216LL), 256, 0, s>>>(B4_, 16777216LL);
  for (int x2 = 0; x2 < 16; ++x2) {
    // Zst1[(a,c),y] = EL1p @ Tdp[x2]
    gemm(s, B2_, Z_ + (size_t)x2 * 65536, B3_, 65536, 256, 256, 256, 256, 256,
         0, 0, 0);
    // EL3[c] += Zst1[:,c,:]^T @ CE[:,x2,:]   (batched over c, accumulate)
    gemm(s, B3_, RP_ + (size_t)x2 * 256, B4_, 256, 256, 256, 65536, 4096, 256,
         1, 0, 1, 256, 256, 0, 65536);
  }
  // Rho4 and loss
  rho4_k<<<256, 256, 0, s>>>(B4_, G_);
  loss_k<<<1, 256, 0, s>>>(G_, H_in, out);
}
